// MultiHeadALiBi_76278619176978
// MI455X (gfx1250) — matrix operand order repaired and checked by reading, not compiled, not run on hardware
//
#include <hip/hip_runtime.h>
#include <hip/hip_bf16.h>
#include <stdint.h>

// Problem constants (match reference)
#define B_ 4
#define S_ 2048
#define E_ 512
#define H_ 8
#define D_ 64

typedef __attribute__((ext_vector_type(16))) _Float16 v16h;
typedef __attribute__((ext_vector_type(8)))  float    v8f;

__device__ __forceinline__ v8f vzero8() {
  v8f z = {0.f, 0.f, 0.f, 0.f, 0.f, 0.f, 0.f, 0.f};
  return z;
}

__device__ __forceinline__ v8f wmma16x16x32(v16h a, v16h b, v8f c) {
  // D = A(16x32,f16) * B(32x16,f16) + C(16x16,f32)
  return __builtin_amdgcn_wmma_f32_16x16x32_f16(
      false, a, false, b, (short)0, c, false, false);
}

// 16-bit A-matrix (16x32) lane layout (contiguous K, matching B):
// K = idx + 16*(lane>=16).  Row m = lane & 15.
__device__ __forceinline__ int a_k_of(int lane, int idx) {
  return idx + ((lane & 16) ? 16 : 0);
}
// 16-bit B-matrix (32x16): lane n = lane&15, K = idx + 16*(lane>=16)
// 32-bit C/D (16x16): VGPR r -> row r + 8*(lane>=16), col = lane&15

// ---------------------------------------------------------------------------
// Kernel 1: per-head projection  Y[b,h,s,d] = sum_e X[b,s,e] * W[h,e,d] + bias
// 8 waves per block share one (b,h).  The 32x64 W k-slab is staged into LDS
// PRE-SWIZZLED into B-fragment order ([nt][lane][idx]) so each lane reads its
// fragment as one aligned v16h (2x ds_load_b128).  The slab is DOUBLE
// BUFFERED: slab k+1 is staged while WMMAs consume slab k (1 barrier/step),
// and all 4 B-fragments are preloaded before the 4-WMMA chain.
// transpose_out=1 stores Y as [B,H,D,S] (for V, so PV B-frags are contiguous).
// ---------------------------------------------------------------------------
__global__ __launch_bounds__(256) void proj_kernel(
    const float* __restrict__ X, const float* __restrict__ W,
    const float* __restrict__ bias, _Float16* __restrict__ Y,
    int transpose_out) {
  // two slabs in B-fragment order: [buf][nt][lane][idx]
  __shared__ __align__(32) _Float16 Bl[2][4 * 32 * 16];

  const int t = threadIdx.x;
  const int wv = t >> 5, lane = t & 31;
  const int bh = blockIdx.x >> 4;   // 32 (b,h) pairs
  const int sblk = blockIdx.x & 15; // 16 s-blocks of 8 tiles
  const int b = bh / H_, h = bh % H_;
  const int s0 = (sblk * 8 + wv) * 16;
  const int m = lane & 15;
  const int n16 = lane & 15;

  v8f acc[4];
#pragma unroll
  for (int i = 0; i < 4; ++i) acc[i] = vzero8();

  const float* xrow = X + ((size_t)b * S_ + s0 + m) * E_;
  const float* wbase = W + (size_t)h * E_ * D_;

  const int li = t * 8;
  const int lK = li >> 6, lN = li & 63;      // K row 0..31, N col base (mult of 8)
  const int s_idx = lK & 15;                 // fragment idx for this K
  const int s_hi = (lK & 16) ? 16 : 0;       // upper-half-wave lanes

  // stage slab for k0=0 into buffer 0
  {
    const float* src = wbase + (size_t)lK * D_ + lN;
#pragma unroll
    for (int j = 0; j < 8; ++j) {
      const int N = lN + j;
      Bl[0][(((N >> 4) * 32 + (N & 15) + s_hi) << 4) + s_idx] =
          (_Float16)src[j];
    }
  }
  __syncthreads();

  for (int k0 = 0; k0 < E_; k0 += 32) {
    const int cur = (k0 >> 5) & 1;
    // stage next slab into the other buffer while computing this one
    if (k0 + 32 < E_) {
      const float* src = wbase + (size_t)(k0 + 32 + lK) * D_ + lN;
      __builtin_prefetch(wbase + (size_t)(k0 + 64 + lK) * D_ + lN, 0, 1);
#pragma unroll
      for (int j = 0; j < 8; ++j) {
        const int N = lN + j;
        Bl[cur ^ 1][(((N >> 4) * 32 + (N & 15) + s_hi) << 4) + s_idx] =
            (_Float16)src[j];
      }
    }

    v16h a;
#pragma unroll
    for (int idx = 0; idx < 16; ++idx)
      a[idx] = (_Float16)xrow[k0 + a_k_of(lane, idx)];

    const v16h* blp = (const v16h*)Bl[cur];
    v16h bf0 = blp[0 * 32 + lane];  // preload all 4 fragments, then WMMA chain
    v16h bf1 = blp[1 * 32 + lane];
    v16h bf2 = blp[2 * 32 + lane];
    v16h bf3 = blp[3 * 32 + lane];
    acc[0] = wmma16x16x32(a, bf0, acc[0]);
    acc[1] = wmma16x16x32(a, bf1, acc[1]);
    acc[2] = wmma16x16x32(a, bf2, acc[2]);
    acc[3] = wmma16x16x32(a, bf3, acc[3]);
    __syncthreads();
  }

#pragma unroll
  for (int nt = 0; nt < 4; ++nt) {
    const int n = nt * 16 + n16;
    const float bval = bias[h * D_ + n];
#pragma unroll
    for (int r = 0; r < 8; ++r) {
      const int mrow = r + ((lane & 16) ? 8 : 0);
      const float vv = acc[nt][r] + bval;
      size_t off;
      if (transpose_out)
        off = ((size_t)(b * H_ + h) * D_ + n) * S_ + (s0 + mrow);
      else
        off = ((size_t)(b * H_ + h) * S_ + (s0 + mrow)) * D_ + n;
      Y[off] = (_Float16)vv;
    }
  }
}

// ---------------------------------------------------------------------------
// Kernel 2: fused attention for one (b, h, 16-row q tile).
// QK^T logits strip [16 x 2048] lives in LDS (f32).  After block-wide softmax
// statistics, each PV lane builds its A-fragment DIRECTLY IN REGISTERS:
// it loads the f32 logit pairs it owns (ds_load_b64), applies exp/Z, the
// distance bias and both masks inline, and feeds WMMA.  No intermediate
// masked-weight buffer, no strided LDS gather.
// ---------------------------------------------------------------------------
#define LG_STRIDE 2052  // f32 elements per logits row (2048 + pad)

__global__ __launch_bounds__(256) void attn_kernel(
    const _Float16* __restrict__ qh, const _Float16* __restrict__ kh,
    const _Float16* __restrict__ vhT, const float* __restrict__ coords_q,
    const float* __restrict__ coords_k, const uint8_t* __restrict__ attn_mask,
    const uint8_t* __restrict__ alibi_mask,
    const float* __restrict__ bias_scale, const float* __restrict__ running_mean,
    _Float16* __restrict__ att) {
  extern __shared__ char smem[];
  float* lgts = (float*)smem;                           // 16*2052*4 = 131328 B
  float* outacc = (float*)(smem + 16 * LG_STRIDE * 4);  // 1024 f32
  float* rowred = outacc + 1024;                        // 256 f32
  float* rowmax = rowred + 256;                         // 16
  float* rowsum = rowmax + 16;                          // 16
  float* cq = rowsum + 16;                              // 48

  const int t = threadIdx.x;
  const int wv = t >> 5, lane = t & 31;
  const int qt = blockIdx.x & 127;
  const int bh = blockIdx.x >> 7;
  const int b = bh >> 3, h = bh & 7;
  const int q0 = qt * 16;
  const int n16 = lane & 15;
  const int kb = (lane & 16) ? 16 : 0;
  const int hi8 = (lane & 16) ? 8 : 0;

  const float sc = bias_scale[h] / running_mean[h];

  // init: q-tile coords + output accumulator
  if (t < 48) cq[t] = coords_q[((size_t)b * S_ + q0 + (t / 3)) * 3 + (t % 3)];
  for (int i = t; i < 1024; i += 256) outacc[i] = 0.f;

  // ---- Phase 1: logits = (qh . kh^T) * D^-0.5, strip -> LDS (f32) ----
  const _Float16* qrow = qh + ((size_t)bh * S_ + q0 + n16) * D_;
  v16h a0, a1;
#pragma unroll
  for (int idx = 0; idx < 16; ++idx) {
    a0[idx] = qrow[a_k_of(lane, idx)];
    a1[idx] = qrow[32 + a_k_of(lane, idx)];
  }
  // 2x unrolled over column tiles: issue 4 fragment loads ahead of WMMAs
  for (int ct = 0; ct < 16; ct += 2) {
    const int colA = wv * 256 + ct * 16;
    const int colB = colA + 16;
    const _Float16* krowA = kh + ((size_t)bh * S_ + colA + n16) * D_ + kb;
    const _Float16* krowB = kh + ((size_t)bh * S_ + colB + n16) * D_ + kb;
    v16h b0A = *(const v16h*)(krowA);
    v16h b1A = *(const v16h*)(krowA + 32);
    v16h b0B = *(const v16h*)(krowB);
    v16h b1B = *(const v16h*)(krowB + 32);
    v8f cA = vzero8(), cB = vzero8();
    cA = wmma16x16x32(a0, b0A, cA);
    cA = wmma16x16x32(a1, b1A, cA);
    cB = wmma16x16x32(a0, b0B, cB);
    cB = wmma16x16x32(a1, b1B, cB);
#pragma unroll
    for (int r = 0; r < 8; ++r) {
      lgts[(r + hi8) * LG_STRIDE + colA + n16] = cA[r] * 0.125f;  // 1/sqrt(64)
      lgts[(r + hi8) * LG_STRIDE + colB + n16] = cB[r] * 0.125f;
    }
  }
  __syncthreads();

  // ---- Phase 2: row max (float4 scans) ----
  const int rr = t >> 4, seg = t & 15;
  {
    const float4* lr4 = (const float4*)(lgts + rr * LG_STRIDE + seg * 128);
    float mx = -3.4e38f;
    for (int i = 0; i < 32; ++i) {
      float4 x = lr4[i];
      mx = fmaxf(mx, fmaxf(fmaxf(x.x, x.y), fmaxf(x.z, x.w)));
    }
    rowred[rr * 16 + seg] = mx;
  }
  __syncthreads();
  if (t < 16) {
    float mx = rowred[t * 16];
    for (int i = 1; i < 16; ++i) mx = fmaxf(mx, rowred[t * 16 + i]);
    rowmax[t] = mx;
  }
  __syncthreads();

  // ---- Phase 3: row sum of exp (float4 scans) ----
  {
    const float mx = rowmax[rr];
    const float4* lr4 = (const float4*)(lgts + rr * LG_STRIDE + seg * 128);
    float sm = 0.f;
    for (int i = 0; i < 32; ++i) {
      float4 x = lr4[i];
      sm += __expf(x.x - mx) + __expf(x.y - mx) + __expf(x.z - mx) +
            __expf(x.w - mx);
    }
    rowred[rr * 16 + seg] = sm;
  }
  __syncthreads();
  if (t < 16) {
    float sm = 0.f;
    for (int i = 0; i < 16; ++i) sm += rowred[t * 16 + i];
    rowsum[t] = sm;
  }
  __syncthreads();

  // ---- Phase 4+5 fused: build masked A-fragments in registers, PV WMMA ----
  {
    const int mrow = lane & 15;  // A-fragment row owned by this lane
    const float mx = rowmax[mrow];
    const float rinv = 1.0f / rowsum[mrow];
    const float cx = cq[mrow * 3 + 0];
    const float cy = cq[mrow * 3 + 1];
    const float cz = cq[mrow * 3 + 2];
    const float* lrow = lgts + mrow * LG_STRIDE;
    const uint8_t* am = attn_mask + ((size_t)b * S_ + q0 + mrow) * S_;
    const uint8_t* al = alibi_mask + ((size_t)b * S_ + q0 + mrow) * S_;
    const float* ckb = coords_k + (size_t)b * S_ * 3;

    v8f acc[4];
#pragma unroll
    for (int i = 0; i < 4; ++i) acc[i] = vzero8();

    for (int ks = 0; ks < 8; ++ks) {
      const int kcol = wv * 256 + ks * 32;
      v16h a;
#pragma unroll
      for (int jj = 0; jj < 8; ++jj) {
        const int kp = a_k_of(lane, 2 * jj);  // even; owns pair (kp, kp+1)
        const int c = kcol + kp;
        const float l0 = lrow[c], l1 = lrow[c + 1];  // ds_load_b64
        const uint16_t am2 = *(const uint16_t*)(am + c);
        const uint16_t al2 = *(const uint16_t*)(al + c);
        const float* ckp = ckb + (size_t)c * 3;
        const float w0 = __expf(l0 - mx) * rinv;
        const float w1 = __expf(l1 - mx) * rinv;
        const float dx0 = cx - ckp[0], dy0 = cy - ckp[1], dz0 = cz - ckp[2];
        const float dx1 = cx - ckp[3], dy1 = cy - ckp[4], dz1 = cz - ckp[5];
        const float d0 = sqrtf(dx0 * dx0 + dy0 * dy0 + dz0 * dz0);
        const float d1 = sqrtf(dx1 * dx1 + dy1 * dy1 + dz1 * dz1);
        const float md0 = (al2 & 0x00ff) ? 0.f : d0 * sc;
        const float md1 = (al2 & 0xff00) ? 0.f : d1 * sc;
        const float v0 = (am2 & 0x00ff) ? 0.f : (w0 - md0);
        const float v1 = (am2 & 0xff00) ? 0.f : (w1 - md1);
        a[2 * jj + 0] = (_Float16)v0;
        a[2 * jj + 1] = (_Float16)v1;
      }
#pragma unroll
      for (int nt = 0; nt < 4; ++nt) {
        const _Float16* vrow =
            vhT + ((size_t)bh * D_ + nt * 16 + n16) * S_ + kcol + kb;
        v16h bf = *(const v16h*)vrow;  // contiguous along K (V pre-transposed)
        acc[nt] = wmma16x16x32(a, bf, acc[nt]);
      }
    }
#pragma unroll
    for (int nt = 0; nt < 4; ++nt)
#pragma unroll
      for (int r = 0; r < 8; ++r)
        atomicAdd(&outacc[(r + hi8) * 64 + nt * 16 + n16], acc[nt][r]);
  }
  __syncthreads();

  // ---- Phase 6: store head tile into [B,S,E] f16 (4 consecutive per thread)
  {
    const int i0 = t * 4;
    const int r = i0 >> 6, d = i0 & 63;
    _Float16* dst = att + ((size_t)b * S_ + q0 + r) * E_ + h * D_ + d;
#pragma unroll
    for (int j = 0; j < 4; ++j) dst[j] = (_Float16)outacc[i0 + j];
  }
}

// ---------------------------------------------------------------------------
// Kernel 3: out = att[B*S, E] . fc_w^T + fc_b  (fc_w rows contiguous along K)
// ---------------------------------------------------------------------------
__global__ __launch_bounds__(256) void outproj_kernel(
    const _Float16* __restrict__ att, const float* __restrict__ fc_w,
    const float* __restrict__ fc_b, float* __restrict__ out) {
  const int t = threadIdx.x;
  const int wv = t >> 5, lane = t & 31;
  const int m0 = (blockIdx.x * 8 + wv) * 16;
  const int n0 = blockIdx.y * 64;
  const int n16 = lane & 15;
  const int kb = (lane & 16) ? 16 : 0;

  v8f acc[4];
#pragma unroll
  for (int i = 0; i < 4; ++i) acc[i] = vzero8();

  const _Float16* arow = att + (size_t)(m0 + (lane & 15)) * E_;

  for (int k0 = 0; k0 < E_; k0 += 32) {
    v16h a;
#pragma unroll
    for (int idx = 0; idx < 16; ++idx)
      a[idx] = arow[k0 + a_k_of(lane, idx)];
#pragma unroll
    for (int nt = 0; nt < 4; ++nt) {
      const float* fwrow = fc_w + (size_t)(n0 + nt * 16 + n16) * E_ + k0 + kb;
      v16h bf;
#pragma unroll
      for (int idx = 0; idx < 16; ++idx) bf[idx] = (_Float16)fwrow[idx];
      acc[nt] = wmma16x16x32(a, bf, acc[nt]);
    }
  }

#pragma unroll
  for (int nt = 0; nt < 4; ++nt) {
    const float bv = fc_b[n0 + nt * 16 + n16];
#pragma unroll
    for (int r = 0; r < 8; ++r) {
      const int mrow = r + ((lane & 16) ? 8 : 0);
      out[(size_t)(m0 + mrow) * E_ + n0 + nt * 16 + n16] = acc[nt][r] + bv;
    }
  }
}

// ---------------------------------------------------------------------------
extern "C" void kernel_launch(void* const* d_in, const int* in_sizes, int n_in,
                              void* d_out, int out_size, void* d_ws,
                              size_t ws_size, hipStream_t stream) {
  (void)in_sizes; (void)n_in; (void)out_size; (void)ws_size;
  const float*   q      = (const float*)d_in[0];
  const float*   k      = (const float*)d_in[1];
  const float*   v      = (const float*)d_in[2];
  const float*   cq     = (const float*)d_in[3];
  const float*   ck     = (const float*)d_in[4];
  const uint8_t* amask  = (const uint8_t*)d_in[5];
  const uint8_t* almask = (const uint8_t*)d_in[6];
  const float*   Wq     = (const float*)d_in[7];
  const float*   bq     = (const float*)d_in[8];
  const float*   Wk     = (const float*)d_in[9];
  const float*   bk     = (const float*)d_in[10];
  const float*   Wv     = (const float*)d_in[11];
  const float*   bv     = (const float*)d_in[12];
  const float*   bsc    = (const float*)d_in[13];
  const float*   rmean  = (const float*)d_in[14];
  const float*   fcw    = (const float*)d_in[15];
  const float*   fcb    = (const float*)d_in[16];

  const size_t HSD = (size_t)B_ * H_ * S_ * D_;  // 4,194,304 elements
  _Float16* qh  = (_Float16*)d_ws;
  _Float16* kh  = qh + HSD;
  _Float16* vhT = kh + HSD;   // stored [B,H,D,S]
  _Float16* att = vhT + HSD;  // stored [B,S,E] f16

  // projections: grid = (B*H pairs) * 16 s-blocks = 512 blocks of 8 waves
  proj_kernel<<<512, 256, 0, stream>>>(q, Wq, bq, qh, 0);
  proj_kernel<<<512, 256, 0, stream>>>(k, Wk, bk, kh, 0);
  proj_kernel<<<512, 256, 0, stream>>>(v, Wv, bv, vhT, 1);

  // attention: one block per (b,h,q-tile) = 4*8*128 = 4096 blocks
  const size_t smem_bytes =
      (size_t)16 * LG_STRIDE * 4 + 1024 * 4 + 256 * 4 + 16 * 4 + 16 * 4 + 48 * 4;
  attn_kernel<<<4096, 256, smem_bytes, stream>>>(qh, kh, vhT, cq, ck, amask,
                                                 almask, bsc, rmean, att);

  // output projection: 512 m-tiles / 8 waves = 64 blocks, 8 n-blocks
  outproj_kernel<<<dim3(64, 8), 256, 0, stream>>>(att, fcw, fcb, (float*)d_out);
}
